// MultiheadAttention_30897994727427
// MI455X (gfx1250) — matrix operand order repaired and checked by reading, not compiled, not run on hardware
//
#include <hip/hip_runtime.h>

// Multi-head attention forward for gfx1250 (MI455X), bf16 WMMA pipeline.
// B=4, L=2048, H=16, D_MODEL=1024, D_QKV=64.
// Round 4: v_perm-based packed bf16 conversion; K stored natural / V stored
// transposed so both attention B-fragment stages are contiguous b128 loads.

#define B_   4
#define L_   2048
#define H_   16
#define DM_  1024
#define DQ_  64
#define MROWS (B_ * L_)          // 8192

typedef __attribute__((ext_vector_type(16))) __bf16          v16bf;
typedef __attribute__((ext_vector_type(16))) unsigned short  v16us;
typedef __attribute__((ext_vector_type(8)))  unsigned short  us8;
typedef __attribute__((ext_vector_type(4)))  unsigned int    u32x4;
typedef __attribute__((ext_vector_type(8)))  float           v8f;

// Convert two f32 -> packed bf16 pair in one u32 (a -> low 16, b -> high 16).
// Round-half-up + v_perm_b32: 3 VALU ops per two elements.
__device__ __forceinline__ unsigned int pk2bf(float a, float b) {
  const unsigned int ua = __float_as_uint(a) + 0x8000u;
  const unsigned int ub = __float_as_uint(b) + 0x8000u;
  // bytes 4-7 = ub, bytes 0-3 = ua; select {ub[3],ub[2],ua[3],ua[2]}
  return __builtin_amdgcn_perm(ub, ua, 0x07060302u);
}

// Pack 8 floats -> 8 bf16 (one 16-byte vector) with packed converts.
__device__ __forceinline__ us8 pack8(const float* x) {
  u32x4 w;
#pragma unroll
  for (int e = 0; e < 4; ++e) w[e] = pk2bf(x[2 * e], x[2 * e + 1]);
  return __builtin_bit_cast(us8, w);
}

__device__ __forceinline__ v8f wmma_bf16(v16us a, v16us b, v8f c) {
  // D = A(16x32 bf16) x B(32x16 bf16) + C(16x16 f32)
  return __builtin_amdgcn_wmma_f32_16x16x32_bf16(
      false, __builtin_bit_cast(v16bf, a),
      false, __builtin_bit_cast(v16bf, b),
      (short)0, c, false, false);
}

// A-fragment (16x32, 16-bit) from row-major tile base[row][stride].
// Per lane: two contiguous 8-element runs -> two b128 loads.
// stride must be a multiple of 8 elements (16B alignment of runs).
__device__ __forceinline__ v16us load_a_frag(const unsigned short* base, int stride) {
  const int lane = threadIdx.x & 31;
  const unsigned short* p = base + (lane & 15) * stride + (lane >> 4) * 16;
  union { v16us v; us8 h[2]; } u;
  u.h[0] = *(const us8*)(p);        // k = hf*16 .. +7
  u.h[1] = *(const us8*)(p + 8);    // k = hf*16 + 8 .. +15
  return u.v;
}

// B-fragment from a fragment-swizzled LDS tile: fragbase[lane][16] contiguous.
// One 32-byte vector load per lane (2x ds_load_b128).
__device__ __forceinline__ v16us load_b_frag_sw(const unsigned short* fragbase) {
  const int lane = threadIdx.x & 31;
  return *(const v16us*)(fragbase + lane * 16);
}

// ---------------------------------------------------------------------------
// Kernel 1: fused QKV projection.  C[r, h*64+d] = sum_m X[r,m] * W[h,m,d]
// Q,K stored [B,H,L,64] (natural); V stored [B,H,64,L] (transposed for PV
// B-fragments). grid = (16, 128, 3), block = 128, tile 64x64, K-step 64.
// ---------------------------------------------------------------------------
__global__ __launch_bounds__(128) void qkv_proj_kernel(
    const float* __restrict__ q, const float* __restrict__ k, const float* __restrict__ v,
    const float* __restrict__ Wq, const float* __restrict__ Wk, const float* __restrict__ Wv,
    unsigned short* __restrict__ Qp, unsigned short* __restrict__ Kp,
    unsigned short* __restrict__ Vp) {
  const int which = blockIdx.z;
  const float* X = (which == 0) ? q : (which == 1) ? k : v;
  const float* W = (which == 0) ? Wq : (which == 1) ? Wk : Wv;
  const int tileN = blockIdx.x;            // == head index (64-wide tiles)
  const int tileM = blockIdx.y;
  const int tid  = threadIdx.x;
  const int wave = tid >> 5;
  const int lane = tid & 31;

  __shared__ __align__(32) unsigned short sA[64][72];        // [m][k] bf16, K-step 64
  __shared__ __align__(32) unsigned short sBs[8][32][16];    // frag-swizzled B (kc*4+nt)

  const v8f zero8 = {0.f, 0.f, 0.f, 0.f, 0.f, 0.f, 0.f, 0.f};
  v8f acc[4] = {zero8, zero8, zero8, zero8};

  for (int kt = 0; kt < DM_ / 64; ++kt) {
    __syncthreads();
    {  // A tile: 64 rows x 64 k, fp32 -> bf16 (packed), vector stores
      const int row = tid >> 1;
      const int c0  = (tid & 1) * 32;
      const float* src = X + (size_t)(tileM * 64 + row) * DM_ + kt * 64 + c0;
      if (kt + 1 < DM_ / 64) __builtin_prefetch(src + 64, 0, 0);
      float vals[32];
#pragma unroll
      for (int e = 0; e < 32; ++e) vals[e] = src[e];
#pragma unroll
      for (int t = 0; t < 4; ++t) *(us8*)&sA[row][c0 + t * 8] = pack8(vals + t * 8);
    }
    {  // B tile, fragment-swizzled; W element (mk, n) strided by DQ in k.
      const int n    = tid & 63;
      const int rsel = tid >> 6;
      const float* wbase = W + (size_t)tileN * DM_ * DQ_ + n;
#pragma unroll
      for (int j = 0; j < 2; ++j) {
        const int r = rsel * 2 + j;          // 16-k run index (0..3)
        const float* src = wbase + (size_t)(kt * 64 + r * 16) * DQ_;
        float vals[16];
#pragma unroll
        for (int e = 0; e < 16; ++e) vals[e] = src[(size_t)e * DQ_];
        const int f = (r >> 1) * 4 + (n >> 4);
        unsigned short* dst = &sBs[f][(n & 15) + (r & 1) * 16][0];
        *(us8*)dst       = pack8(vals);
        *(us8*)(dst + 8) = pack8(vals + 8);
      }
    }
    __syncthreads();

#pragma unroll
    for (int kc = 0; kc < 2; ++kc) {
      const v16us a = load_a_frag(&sA[wave * 16][kc * 32], 72);
#pragma unroll
      for (int nt = 0; nt < 4; ++nt) {
        acc[nt] = wmma_bf16(a, load_b_frag_sw(&sBs[kc * 4 + nt][0][0]), acc[nt]);
      }
    }
  }

  // Epilogue: C layout -> bf16 stores (uniform branch hoisted).
  const int n  = lane & 15;
  const int hf = lane >> 4;
  if (which != 2) {  // Q and K: natural [B,H,L,64]
    unsigned short* dst = (which == 0) ? Qp : Kp;
#pragma unroll
    for (int nt = 0; nt < 4; ++nt) {
#pragma unroll
      for (int i = 0; i < 8; i += 2) {
        const unsigned int pw = pk2bf(acc[nt][i], acc[nt][i + 1]);
#pragma unroll
        for (int s = 0; s < 2; ++s) {
          const int r  = tileM * 64 + wave * 16 + i + s + hf * 8;
          const int bb = r >> 11;
          const int l  = r & (L_ - 1);
          dst[((size_t)(bb * H_ + tileN) * L_ + l) * DQ_ + nt * 16 + n] =
              (unsigned short)(s ? (pw >> 16) : (pw & 0xFFFF));
        }
      }
    }
  } else {           // V: transposed [B,H,64,L]
#pragma unroll
    for (int nt = 0; nt < 4; ++nt) {
#pragma unroll
      for (int i = 0; i < 8; i += 2) {
        const unsigned int pw = pk2bf(acc[nt][i], acc[nt][i + 1]);
#pragma unroll
        for (int s = 0; s < 2; ++s) {
          const int r  = tileM * 64 + wave * 16 + i + s + hf * 8;
          const int bb = r >> 11;
          const int l  = r & (L_ - 1);
          Vp[((size_t)(bb * H_ + tileN) * DQ_ + nt * 16 + n) * L_ + l] =
              (unsigned short)(s ? (pw >> 16) : (pw & 0xFFFF));
        }
      }
    }
  }
}

// ---------------------------------------------------------------------------
// Kernel 2: flash attention per (b,h). Block = 64 queries (4 waves x 16),
// 32-key slabs: S = Q K^T (4 WMMA), mask, online softmax, O += P V (4 WMMA).
// K/V fragment staging is contiguous b128 global -> b128 LDS.
// grid = (L/64, B*H), block = 128.
// ---------------------------------------------------------------------------
__global__ __launch_bounds__(128) void attention_kernel(
    const unsigned short* __restrict__ Qp, const unsigned short* __restrict__ Kp,
    const unsigned short* __restrict__ Vp, const int* __restrict__ mask,
    unsigned short* __restrict__ AO) {
  const int bh = blockIdx.y;
  const int b  = bh >> 4;
  const int h  = bh & (H_ - 1);
  const int tid  = threadIdx.x;
  const int wave = tid >> 5;
  const int lane = tid & 31;
  const int m  = lane & 15;
  const int hf = lane >> 4;
  const int q0 = blockIdx.x * 64 + wave * 16;

  __shared__ __align__(32) unsigned short sKs[4][32][16];   // frag f = kk*2+nt
  __shared__ __align__(32) unsigned short sVs[4][32][16];   // frag f = nt2
  __shared__ __align__(32) unsigned short sP[4][16][40];    // per-wave probabilities

  // Q A-fragments resident in VGPRs: two b128 loads per 32-d chunk.
  const unsigned short* qrow = Qp + ((size_t)bh * L_ + q0 + m) * DQ_;
  v16us qa[2];
#pragma unroll
  for (int kk = 0; kk < 2; ++kk) {
    const unsigned short* p = qrow + kk * 32 + hf * 16;
    union { v16us v; us8 hv[2]; } u;
    u.hv[0] = *(const us8*)(p);
    u.hv[1] = *(const us8*)(p + 8);
    qa[kk] = u.v;
  }

  const v8f zero8 = {0.f, 0.f, 0.f, 0.f, 0.f, 0.f, 0.f, 0.f};
  v8f o[4] = {zero8, zero8, zero8, zero8};
  float rm[8], rs[8];
#pragma unroll
  for (int i = 0; i < 8; ++i) { rm[i] = -3.0e38f; rs[i] = 0.f; }

  const unsigned short* Kbase = Kp + (size_t)bh * L_ * DQ_;   // [key][d]
  const unsigned short* Vbase = Vp + (size_t)bh * DQ_ * L_;   // [d][key]
  const int* mbase = mask + (size_t)b * L_ * L_;
  const int n = lane & 15;

  for (int kt = 0; kt < L_ / 32; ++kt) {
    const int k0 = kt * 32;
    __syncthreads();
    {  // K fragments: 16-d run at fixed key -> contiguous in [key][d].
      const int f  = tid >> 5;               // 0..3 : kk*2 + nt
      const int l  = tid & 31;
      const int kk = f >> 1;
      const int nt = f & 1;
      const int key = k0 + nt * 16 + (l & 15);
      const int d0  = kk * 32 + (l >> 4) * 16;
      const unsigned short* src = Kbase + (size_t)key * DQ_ + d0;
      us8 r0 = *(const us8*)(src);
      us8 r1 = *(const us8*)(src + 8);
      unsigned short* dst = &sKs[f][l][0];
      *(us8*)dst       = r0;
      *(us8*)(dst + 8) = r1;
    }
    {  // V fragments: 16-key run at fixed d -> contiguous in [d][key].
      const int f = tid >> 5;                // nt2
      const int l = tid & 31;
      const int d   = f * 16 + (l & 15);
      const int kr0 = k0 + (l >> 4) * 16;
      const unsigned short* src = Vbase + (size_t)d * L_ + kr0;
      us8 r0 = *(const us8*)(src);
      us8 r1 = *(const us8*)(src + 8);
      unsigned short* dst = &sVs[f][l][0];
      *(us8*)dst       = r0;
      *(us8*)(dst + 8) = r1;
    }
    __syncthreads();

    // S = Q K^T for 16q x 32k : two 16x16 tiles, k-dim 64 -> 2 steps each.
    v8f s[2] = {zero8, zero8};
#pragma unroll
    for (int nt = 0; nt < 2; ++nt) {
#pragma unroll
      for (int kk = 0; kk < 2; ++kk) {
        s[nt] = wmma_bf16(qa[kk], load_b_frag_sw(&sKs[kk * 2 + nt][0][0]), s[nt]);
      }
    }

    // Online softmax over this 32-key slab.
#pragma unroll
    for (int i = 0; i < 8; ++i) {
      const int qq = q0 + i + hf * 8;
      float s0 = s[0][i] * 0.125f;   // 1/sqrt(64)
      float s1 = s[1][i] * 0.125f;
      const int m0 = mbase[(size_t)qq * L_ + k0 + n];
      const int m1 = mbase[(size_t)qq * L_ + k0 + 16 + n];
      s0 = m0 ? s0 : -1.0e9f;
      s1 = m1 ? s1 : -1.0e9f;
      float tmax = fmaxf(s0, s1);
#pragma unroll
      for (int off = 1; off < 16; off <<= 1)
        tmax = fmaxf(tmax, __shfl_xor(tmax, off, 32));   // within 16-lane half
      const float nm   = fmaxf(rm[i], tmax);
      const float corr = __expf(rm[i] - nm);
      rm[i] = nm;
      const float p0 = __expf(s0 - nm);
      const float p1 = __expf(s1 - nm);
      float psum = p0 + p1;
#pragma unroll
      for (int off = 1; off < 16; off <<= 1)
        psum += __shfl_xor(psum, off, 32);
      rs[i] = rs[i] * corr + psum;
#pragma unroll
      for (int nt2 = 0; nt2 < 4; ++nt2) o[nt2][i] *= corr;
      const unsigned int pw = pk2bf(p0, p1);
      sP[wave][i + hf * 8][n]      = (unsigned short)(pw & 0xFFFF);
      sP[wave][i + hf * 8][16 + n] = (unsigned short)(pw >> 16);
    }
    asm volatile("" ::: "memory");  // keep P stores before re-layout loads

    // O += P(16x32) x V(32x64). Same-wave LDS RAW is in-order on CDNA5.
    const v16us pa = load_a_frag(&sP[wave][0][0], 40);
#pragma unroll
    for (int nt2 = 0; nt2 < 4; ++nt2) {
      o[nt2] = wmma_bf16(pa, load_b_frag_sw(&sVs[nt2][0][0]), o[nt2]);
    }
  }

  // Epilogue: normalize and store concat-head layout [B*L, 1024] as bf16.
#pragma unroll
  for (int nt2 = 0; nt2 < 4; ++nt2) {
#pragma unroll
    for (int i = 0; i < 8; i += 2) {
      const unsigned int pw = pk2bf(o[nt2][i] / rs[i], o[nt2][i + 1] / rs[i + 1]);
#pragma unroll
      for (int s = 0; s < 2; ++s) {
        const int qq = q0 + i + s + hf * 8;
        AO[((size_t)b * L_ + qq) * DM_ + h * DQ_ + nt2 * 16 + n] =
            (unsigned short)(s ? (pw >> 16) : (pw & 0xFFFF));
      }
    }
  }
}

// ---------------------------------------------------------------------------
// Kernel 3: output projection  out = AO @ Wo^T + bias  (fp32 out).
// grid = (16, 128), block = 128, tile 64x64, K-step 64.
// ---------------------------------------------------------------------------
__global__ __launch_bounds__(128) void out_proj_kernel(
    const unsigned short* __restrict__ AO, const float* __restrict__ Wo,
    const float* __restrict__ bias, float* __restrict__ out) {
  const int tileN = blockIdx.x;
  const int tileM = blockIdx.y;
  const int tid  = threadIdx.x;
  const int wave = tid >> 5;
  const int lane = tid & 31;

  __shared__ __align__(32) unsigned short sA[64][72];
  __shared__ __align__(32) unsigned short sBs[8][32][16];

  const v8f zero8 = {0.f, 0.f, 0.f, 0.f, 0.f, 0.f, 0.f, 0.f};
  v8f acc[4] = {zero8, zero8, zero8, zero8};

  for (int kt = 0; kt < DM_ / 64; ++kt) {
    __syncthreads();
    {  // A tile (already bf16): 64B contiguous per thread -> b128 in/out
      const int row = tid >> 1;
      const int c0  = (tid & 1) * 32;
      const unsigned short* src = AO + (size_t)(tileM * 64 + row) * DM_ + kt * 64 + c0;
#pragma unroll
      for (int t = 0; t < 4; ++t)
        *(us8*)&sA[row][c0 + t * 8] = *(const us8*)(src + t * 8);
    }
    {  // B tile, fragment-swizzled: B(k,n) = Wo[n_glob*DM + k_glob], contiguous in k.
      const int n    = tid & 63;
      const int rsel = tid >> 6;
      const float* src = Wo + (size_t)(tileN * 64 + n) * DM_ + kt * 64 + rsel * 32;
      if (kt + 1 < DM_ / 64) __builtin_prefetch(src + 64, 0, 0);
      float vals[32];
#pragma unroll
      for (int e = 0; e < 32; ++e) vals[e] = src[e];
      const int f = rsel * 4 + (n >> 4);
#pragma unroll
      for (int j = 0; j < 2; ++j) {
        unsigned short* dst = &sBs[f][(n & 15) + j * 16][0];
        *(us8*)dst       = pack8(vals + j * 16);
        *(us8*)(dst + 8) = pack8(vals + j * 16 + 8);
      }
    }
    __syncthreads();

#pragma unroll
    for (int kc = 0; kc < 2; ++kc) {
      const v16us a = load_a_frag(&sA[wave * 16][kc * 32], 72);
#pragma unroll
      for (int nt = 0; nt < 4; ++nt) {
        acc[nt] = wmma_bf16(a, load_b_frag_sw(&sBs[kc * 4 + nt][0][0]), acc[nt]);
      }
    }
  }

  const int n  = lane & 15;
  const int hf = lane >> 4;
#pragma unroll
  for (int nt = 0; nt < 4; ++nt) {
#pragma unroll
    for (int i = 0; i < 8; ++i) {
      const int r = tileM * 64 + wave * 16 + i + hf * 8;
      const int j = tileN * 64 + nt * 16 + n;
      out[(size_t)r * DM_ + j] = acc[nt][i] + bias[j];
    }
  }
}

// ---------------------------------------------------------------------------
extern "C" void kernel_launch(void* const* d_in, const int* in_sizes, int n_in,
                              void* d_out, int out_size, void* d_ws, size_t ws_size,
                              hipStream_t stream) {
  (void)in_sizes; (void)n_in; (void)out_size; (void)ws_size;
  const float* q    = (const float*)d_in[0];
  const float* k    = (const float*)d_in[1];
  const float* v    = (const float*)d_in[2];
  const int*   mask = (const int*)d_in[3];
  const float* Wq   = (const float*)d_in[4];
  const float* Wk   = (const float*)d_in[5];
  const float* Wv   = (const float*)d_in[6];
  const float* Wo_w = (const float*)d_in[7];
  const float* Wo_b = (const float*)d_in[8];
  float* out = (float*)d_out;

  // Workspace: bf16 Q,K ([B,H,L,64]), V ([B,H,64,L]), AO ([B*L,1024])
  const size_t SEG = (size_t)B_ * H_ * L_ * DQ_ * sizeof(unsigned short);  // 16 MB
  char* ws = (char*)d_ws;
  unsigned short* Qp = (unsigned short*)(ws);
  unsigned short* Kp = (unsigned short*)(ws + SEG);
  unsigned short* Vp = (unsigned short*)(ws + 2 * SEG);
  unsigned short* AO = (unsigned short*)(ws + 3 * SEG);

  qkv_proj_kernel<<<dim3(DM_ / 64, MROWS / 64, 3), 128, 0, stream>>>(
      q, k, v, Wq, Wk, Wv, Qp, Kp, Vp);
  attention_kernel<<<dim3(L_ / 64, B_ * H_), 128, 0, stream>>>(
      Qp, Kp, Vp, mask, AO);
  out_proj_kernel<<<dim3(DM_ / 64, MROWS / 64), 128, 0, stream>>>(
      AO, Wo_w, Wo_b, out);
}
